// GNNModel_10007273799836
// MI455X (gfx1250) — compile-verified, matrix operand order checked by reading, not run on hardware
//
#include <hip/hip_runtime.h>

// Fused EGNN for MI455X (gfx1250, wave32, WMMA).
// Pipeline per launch:
//   1) convert+transpose+pad all MLP weights to f16 (zero-padded)
//   2) embed: feats = x@embedW+b (f32 + f16 copy), coors = pos
//   3) per layer: zero accumulators -> fused edge kernel (WMMA) -> node kernel (WMMA)
//   4) final matvec feats @ linW + linb
// All matmuls use v_wmma_f32_16x16x32_f16; intra-wave LDS staging is ordered
// with s_wait_dscnt (CDNA5 split counters); segment sums via f32 global atomics
// into L2-resident accumulators. SiLU uses v_rcp_f32 (fast sigmoid) to keep the
// VALU epilogue cheap relative to the WMMA pipe.

typedef __attribute__((ext_vector_type(16))) _Float16 v16h;
typedef __attribute__((ext_vector_type(8)))  _Float16 v8h;
typedef __attribute__((ext_vector_type(8)))  float    v8f;

#define NN   50000
#define NE   400000
#define HDIM 64
#define EINP 160     // e_in K padded: 130 -> 160 (5 chunks of 32)
#define E1N  260     // edge MLP hidden
#define E1NP 272     // padded to 17 tiles of 16
#define E2KP 288     // K for edge MLP2 padded: 260 -> 288 (9 chunks of 32)
#define CH1  256     // coors MLP hidden
#define NKP  128     // node MLP K (exact, 4 chunks of 32)
#define WAVES 4

__device__ __forceinline__ float silu_f(float x) {
  // x * sigmoid(x) with v_rcp_f32 instead of the IEEE division fix-up chain
  return x * __builtin_amdgcn_rcpf(1.0f + __expf(-x));
}

// Intra-wave LDS ordering: same-wave DS ops complete in order; waiting DScnt==0
// makes prior lane-stores visible to subsequent cross-lane loads without a
// block barrier.
__device__ __forceinline__ void wave_lds_sync() {
  asm volatile("s_wait_dscnt 0" ::: "memory");
}

// Load a 16x32 f16 fragment (A layout; also used for B == W^T stored row-major
// [N x K]). Per the CDNA5 ISA 16-bit A layout, lane l (m=l&15, kh=l>>4) holds
// K = kh*8 + [0..7] and K = 16 + kh*8 + [0..7]: two contiguous 16B segments.
__device__ __forceinline__ v16h frag_ld(const _Float16* base, int stride, int kc) {
  const int lane = threadIdx.x & 31;
  const int m  = lane & 15;
  const int kb = (lane >> 4) << 3;
  const _Float16* p = base + m * stride + (kc << 5) + kb;
  v8h lo = *(const v8h*)(p);
  v8h hi = *(const v8h*)(p + 16);
  v16h r;
#pragma unroll
  for (int i = 0; i < 8; ++i) { r[i] = lo[i]; r[i + 8] = hi[i]; }
  return r;
}

// ---- weight conversion: f32 [K x R] -> f16 transposed padded [PR x PK], zero pad
__global__ void egnn_convertT(const float* __restrict__ src, _Float16* __restrict__ dst,
                              int R, int K, int PK, int total) {
  int idx = blockIdx.x * blockDim.x + threadIdx.x;
  if (idx >= total) return;
  int nrow = idx / PK;
  int k    = idx - nrow * PK;
  float v = (nrow < R && k < K) ? src[(size_t)k * R + nrow] : 0.0f;
  dst[idx] = (_Float16)v;
}

// ---- embedding + coordinate init
__global__ void egnn_embed(const float* __restrict__ x, const float* __restrict__ pos,
                           const float* __restrict__ eW, const float* __restrict__ eb,
                           float* __restrict__ feats, _Float16* __restrict__ f16,
                           float* __restrict__ coors) {
  const int node = blockIdx.x * blockDim.x + threadIdx.x;
  if (node >= NN) return;
  float xi[8];
#pragma unroll
  for (int k = 0; k < 8; ++k) xi[k] = x[(size_t)node * 8 + k];
  for (int h = 0; h < HDIM; ++h) {
    float s = eb[h];
#pragma unroll
    for (int k = 0; k < 8; ++k) s += xi[k] * eW[k * HDIM + h];
    feats[(size_t)node * HDIM + h] = s;
    f16[(size_t)node * HDIM + h]   = (_Float16)s;
  }
  coors[node * 3 + 0] = pos[node * 3 + 0];
  coors[node * 3 + 1] = pos[node * 3 + 1];
  coors[node * 3 + 2] = pos[node * 3 + 2];
}

// ---- zero the per-layer accumulators
__global__ void egnn_zero(float* __restrict__ m_acc, float* __restrict__ c_acc) {
  int i = blockIdx.x * blockDim.x + threadIdx.x;
  if (i < NN * HDIM) m_acc[i] = 0.0f;
  else if (i < NN * HDIM + NN * 3) c_acc[i - NN * HDIM] = 0.0f;
}

// ---- fused edge kernel: e_in -> edge MLP -> coors MLP -> atomic scatter
__global__ __launch_bounds__(WAVES * 32, 1) void egnn_edge(
    const int* __restrict__ eidx, const float* __restrict__ eattr,
    const float* __restrict__ coors, const _Float16* __restrict__ f16,
    const _Float16* __restrict__ gW1, const _Float16* __restrict__ gW2,
    const _Float16* __restrict__ gCW1,
    const float* __restrict__ eb1, const float* __restrict__ eb2,
    const float* __restrict__ cb1, const float* __restrict__ cw2v,
    const float* __restrict__ cb2,
    float* __restrict__ m_acc, float* __restrict__ c_acc) {
  __shared__ __attribute__((aligned(16))) _Float16 sW1[E1NP * EINP];   // eW1^T padded
  __shared__ __attribute__((aligned(16))) _Float16 sW2[HDIM * E2KP];   // eW2^T padded
  __shared__ __attribute__((aligned(16))) _Float16 sC1[CH1 * HDIM];    // cW1^T
  __shared__ __attribute__((aligned(16))) float    sCw2[CH1];          // cW2 vector
  __shared__ __attribute__((aligned(16))) _Float16 sBuf[WAVES][16 * CH1]; // e_in / c1
  __shared__ __attribute__((aligned(16))) _Float16 sH1[WAVES][16 * E2KP];
  __shared__ __attribute__((aligned(16))) _Float16 sM[WAVES][16 * HDIM];
  __shared__ int sRid[WAVES][16];
  __shared__ int sCid[WAVES][16];

  const int tid = threadIdx.x;
  for (int i = tid; i < (E1NP * EINP) / 8; i += WAVES * 32) ((v8h*)sW1)[i] = ((const v8h*)gW1)[i];
  for (int i = tid; i < (HDIM * E2KP) / 8; i += WAVES * 32) ((v8h*)sW2)[i] = ((const v8h*)gW2)[i];
  for (int i = tid; i < (CH1 * HDIM) / 8; i += WAVES * 32) ((v8h*)sC1)[i] = ((const v8h*)gCW1)[i];
  for (int i = tid; i < CH1; i += WAVES * 32) sCw2[i] = cw2v[i];
  __syncthreads();

  const int lane = tid & 31;
  const int wv   = tid >> 5;
  const int n    = lane & 15;
  const int mb   = (lane >> 4) << 3;
  _Float16* ein = sBuf[wv];
  _Float16* h1  = sH1[wv];
  _Float16* mm  = sM[wv];
  int* rid = sRid[wv];
  int* cid = sCid[wv];

  // zero the K-pad of h1 (cols 272..287) once; never overwritten
  for (int i = lane; i < 256; i += 32)
    h1[(i >> 4) * E2KP + 272 + (i & 15)] = (_Float16)0.0f;

  for (int t = blockIdx.x * WAVES + wv; t < NE / 16; t += gridDim.x * WAVES) {
    float rx = 0.0f, ry = 0.0f, rz = 0.0f;
    if (lane < 16) {
      const int e  = t * 16 + lane;
      const int ri = eidx[e], ci = eidx[NE + e];
      rid[lane] = ri;
      cid[lane] = ci;
      rx = coors[(size_t)ri * 3 + 0] - coors[(size_t)ci * 3 + 0];
      ry = coors[(size_t)ri * 3 + 1] - coors[(size_t)ci * 3 + 1];
      rz = coors[(size_t)ri * 3 + 2] - coors[(size_t)ci * 3 + 2];
      const float dist = rx * rx + ry * ry + rz * rz;
      v8h z = {};
      *(v8h*)(ein + lane * EINP + 128) = z;
      *(v8h*)(ein + lane * EINP + 136) = z;
      *(v8h*)(ein + lane * EINP + 144) = z;
      *(v8h*)(ein + lane * EINP + 152) = z;
      ein[lane * EINP + 128] = (_Float16)eattr[e];
      ein[lane * EINP + 129] = (_Float16)dist;
    }
    wave_lds_sync();
    {  // gather feats16[row] -> cols 0..63, feats16[col] -> cols 64..127
      const int part = lane >> 4;
      const int nid  = part ? cid[n] : rid[n];
      const v8h* src = (const v8h*)(f16 + (size_t)nid * HDIM);
      v8h* dst = (v8h*)(ein + n * EINP + part * 64);
#pragma unroll
      for (int j = 0; j < 8; ++j) dst[j] = src[j];
    }
    wave_lds_sync();

    // edge MLP1: [16 x 160] @ [160 x 272] -> silu -> h1
    for (int nt = 0; nt < 17; ++nt) {
      const int nc  = nt * 16 + n;
      const float b = (nc < E1N) ? eb1[nc] : 0.0f;
      v8f acc = {b, b, b, b, b, b, b, b};
#pragma unroll
      for (int kc = 0; kc < 5; ++kc) {
        v16h a = frag_ld(ein, EINP, kc);
        v16h w = frag_ld(sW1 + nt * 16 * EINP, EINP, kc);
        acc = __builtin_amdgcn_wmma_f32_16x16x32_f16(false, a, false, w, (short)0, acc, false, false);
      }
#pragma unroll
      for (int i = 0; i < 8; ++i)
        h1[(mb + i) * E2KP + nc] = (_Float16)silu_f(acc[i]);
    }
    wave_lds_sync();

    // edge MLP2: [16 x 288] @ [288 x 64] -> silu -> m (f32 regs + f16 LDS)
    v8f mreg[4];
    for (int nt = 0; nt < 4; ++nt) {
      const float b = eb2[nt * 16 + n];
      v8f acc = {b, b, b, b, b, b, b, b};
#pragma unroll
      for (int kc = 0; kc < 9; ++kc) {
        v16h a = frag_ld(h1, E2KP, kc);
        v16h w = frag_ld(sW2 + nt * 16 * E2KP, E2KP, kc);
        acc = __builtin_amdgcn_wmma_f32_16x16x32_f16(false, a, false, w, (short)0, acc, false, false);
      }
#pragma unroll
      for (int i = 0; i < 8; ++i) {
        const float v = silu_f(acc[i]);
        acc[i] = v;
        mm[(mb + i) * HDIM + nt * 16 + n] = (_Float16)v;
      }
      mreg[nt] = acc;
    }
    wave_lds_sync();

    // coors MLP1: [16 x 64] @ [64 x 256] -> silu -> c1 (reuses e_in buffer)
    _Float16* c1 = ein;
    for (int nt = 0; nt < 16; ++nt) {
      const float b = cb1[nt * 16 + n];
      v8f acc = {b, b, b, b, b, b, b, b};
#pragma unroll
      for (int kc = 0; kc < 2; ++kc) {
        v16h a = frag_ld(mm, HDIM, kc);
        v16h w = frag_ld(sC1 + nt * 16 * HDIM, HDIM, kc);
        acc = __builtin_amdgcn_wmma_f32_16x16x32_f16(false, a, false, w, (short)0, acc, false, false);
      }
#pragma unroll
      for (int i = 0; i < 8; ++i)
        c1[(mb + i) * CH1 + nt * 16 + n] = (_Float16)silu_f(acc[i]);
    }
    wave_lds_sync();

    // coors MLP2: per-edge dot [256] -> cw; atomic scatter cw*rel into coor acc
    {
      const int seg = (lane >> 4) * 128;
      const v8h* cv = (const v8h*)(c1 + n * CH1 + seg);
      float s = 0.0f;
#pragma unroll
      for (int kk = 0; kk < 16; ++kk) {
        v8h c = cv[kk];
#pragma unroll
        for (int i = 0; i < 8; ++i)
          s += (float)c[i] * sCw2[seg + kk * 8 + i];
      }
      s += __shfl_xor(s, 16, 32);
      const float cw = s + cb2[0];
      if (lane < 16) {
        const int node = rid[lane];
        atomicAdd(&c_acc[(size_t)node * 3 + 0], cw * rx);
        atomicAdd(&c_acc[(size_t)node * 3 + 1], cw * ry);
        atomicAdd(&c_acc[(size_t)node * 3 + 2], cw * rz);
      }
    }
    // scatter m into m_acc (segment_sum over target nodes)
#pragma unroll
    for (int nt = 0; nt < 4; ++nt) {
#pragma unroll
      for (int i = 0; i < 8; ++i) {
        const int node = rid[mb + i];
        atomicAdd(&m_acc[(size_t)node * HDIM + nt * 16 + n], mreg[nt][i]);
      }
    }
    wave_lds_sync();
  }
}

// ---- node kernel: coors update + node MLP (residual) + f16 refresh
__global__ __launch_bounds__(WAVES * 32, 1) void egnn_node(
    const _Float16* __restrict__ gN1, const _Float16* __restrict__ gN2,
    const float* __restrict__ nb1, const float* __restrict__ nb2,
    const float* __restrict__ m_acc, const float* __restrict__ c_acc,
    float* __restrict__ coors, float* __restrict__ feats, _Float16* __restrict__ f16) {
  __shared__ __attribute__((aligned(16))) _Float16 sN1[128 * NKP];
  __shared__ __attribute__((aligned(16))) _Float16 sN2[HDIM * NKP];
  __shared__ __attribute__((aligned(16))) _Float16 sH[WAVES][16 * NKP];
  __shared__ __attribute__((aligned(16))) _Float16 sG[WAVES][16 * NKP];

  const int tid = threadIdx.x;
  for (int i = tid; i < (128 * NKP) / 8; i += WAVES * 32) ((v8h*)sN1)[i] = ((const v8h*)gN1)[i];
  for (int i = tid; i < (HDIM * NKP) / 8; i += WAVES * 32) ((v8h*)sN2)[i] = ((const v8h*)gN2)[i];
  __syncthreads();

  const int lane = tid & 31;
  const int wv   = tid >> 5;
  const int n    = lane & 15;
  const int mb   = (lane >> 4) << 3;
  _Float16* hb = sH[wv];
  _Float16* h1 = sG[wv];

  for (int t = blockIdx.x * WAVES + wv; t < NN / 16; t += gridDim.x * WAVES) {
    {
      const int part = lane >> 4;
      const int node = t * 16 + n;
      if (part == 0) {  // h cols 0..63 = feats (f16)
        const v8h* src = (const v8h*)(f16 + (size_t)node * HDIM);
        v8h* dst = (v8h*)(hb + n * NKP);
#pragma unroll
        for (int j = 0; j < 8; ++j) dst[j] = src[j];
      } else {          // h cols 64..127 = m_i (f32 -> f16)
        const float4* src = (const float4*)(m_acc + (size_t)node * HDIM);
        _Float16* dst = hb + n * NKP + 64;
#pragma unroll
        for (int j = 0; j < 16; ++j) {
          float4 v = src[j];
          dst[j * 4 + 0] = (_Float16)v.x;
          dst[j * 4 + 1] = (_Float16)v.y;
          dst[j * 4 + 2] = (_Float16)v.z;
          dst[j * 4 + 3] = (_Float16)v.w;
        }
      }
      if (lane < 16) {  // coors += segment_sum(cw*rel)
        coors[(size_t)node * 3 + 0] += c_acc[(size_t)node * 3 + 0];
        coors[(size_t)node * 3 + 1] += c_acc[(size_t)node * 3 + 1];
        coors[(size_t)node * 3 + 2] += c_acc[(size_t)node * 3 + 2];
      }
    }
    wave_lds_sync();

    // node MLP1: [16 x 128] @ [128 x 128] -> silu
    for (int nt = 0; nt < 8; ++nt) {
      const float b = nb1[nt * 16 + n];
      v8f acc = {b, b, b, b, b, b, b, b};
#pragma unroll
      for (int kc = 0; kc < 4; ++kc) {
        v16h a = frag_ld(hb, NKP, kc);
        v16h w = frag_ld(sN1 + nt * 16 * NKP, NKP, kc);
        acc = __builtin_amdgcn_wmma_f32_16x16x32_f16(false, a, false, w, (short)0, acc, false, false);
      }
#pragma unroll
      for (int i = 0; i < 8; ++i)
        h1[(mb + i) * NKP + nt * 16 + n] = (_Float16)silu_f(acc[i]);
    }
    wave_lds_sync();

    // node MLP2: [16 x 128] @ [128 x 64], residual into feats (+f16 copy)
    for (int nt = 0; nt < 4; ++nt) {
      const float b = nb2[nt * 16 + n];
      v8f acc = {b, b, b, b, b, b, b, b};
#pragma unroll
      for (int kc = 0; kc < 4; ++kc) {
        v16h a = frag_ld(h1, NKP, kc);
        v16h w = frag_ld(sN2 + nt * 16 * NKP, NKP, kc);
        acc = __builtin_amdgcn_wmma_f32_16x16x32_f16(false, a, false, w, (short)0, acc, false, false);
      }
#pragma unroll
      for (int i = 0; i < 8; ++i) {
        const size_t idx = (size_t)(t * 16 + mb + i) * HDIM + nt * 16 + n;
        const float nv = feats[idx] + acc[i];
        feats[idx] = nv;
        f16[idx]   = (_Float16)nv;
      }
    }
    wave_lds_sync();
  }
}

// ---- final projection: out = feats @ linW + linb
__global__ void egnn_final(const float* __restrict__ feats, const float* __restrict__ linW,
                           const float* __restrict__ linb, float* __restrict__ out) {
  const int node = blockIdx.x * blockDim.x + threadIdx.x;
  if (node >= NN) return;
  float s = linb[0];
#pragma unroll
  for (int k = 0; k < HDIM; ++k) s += feats[(size_t)node * HDIM + k] * linW[k];
  out[node] = s;
}

extern "C" void kernel_launch(void* const* d_in, const int* in_sizes, int n_in,
                              void* d_out, int out_size, void* d_ws, size_t ws_size,
                              hipStream_t stream) {
  (void)in_sizes; (void)n_in; (void)out_size; (void)ws_size;
  const float* x      = (const float*)d_in[0];
  const float* pos    = (const float*)d_in[1];
  const int*   eidx   = (const int*)d_in[2];
  const float* eattr  = (const float*)d_in[3];
  const float* embedW = (const float*)d_in[4];
  const float* embedb = (const float*)d_in[5];
  const float* eW1    = (const float*)d_in[6];
  const float* eb1    = (const float*)d_in[7];
  const float* eW2    = (const float*)d_in[8];
  const float* eb2    = (const float*)d_in[9];
  const float* cW1    = (const float*)d_in[10];
  const float* cb1    = (const float*)d_in[11];
  const float* cW2    = (const float*)d_in[12];
  const float* cb2    = (const float*)d_in[13];
  const float* nW1    = (const float*)d_in[14];
  const float* nb1    = (const float*)d_in[15];
  const float* nW2    = (const float*)d_in[16];
  const float* nb2    = (const float*)d_in[17];
  const float* linW   = (const float*)d_in[18];
  const float* linb   = (const float*)d_in[19];

  char* base = (char*)d_ws;
  size_t off = 0;
  auto carve = [&](size_t bytes) -> void* {
    void* p = base + off;
    off = (off + bytes + 255) & ~(size_t)255;
    return p;
  };
  float*     feats = (float*)carve((size_t)NN * HDIM * 4);
  _Float16*  f16   = (_Float16*)carve((size_t)NN * HDIM * 2);
  float*     coors = (float*)carve((size_t)NN * 3 * 4);
  float*     m_acc = (float*)carve((size_t)NN * HDIM * 4);
  float*     c_acc = (float*)carve((size_t)NN * 3 * 4);
  _Float16*  tW1   = (_Float16*)carve((size_t)3 * E1NP * EINP * 2);
  _Float16*  tW2   = (_Float16*)carve((size_t)3 * HDIM * E2KP * 2);
  _Float16*  tC1   = (_Float16*)carve((size_t)3 * CH1 * HDIM * 2);
  _Float16*  tN1   = (_Float16*)carve((size_t)3 * 128 * NKP * 2);
  _Float16*  tN2   = (_Float16*)carve((size_t)3 * HDIM * NKP * 2);

  for (int l = 0; l < 3; ++l) {
    auto cvt = [&](const float* src, _Float16* dst, int R, int K, int PR, int PK) {
      const int total = PR * PK;
      egnn_convertT<<<(total + 255) / 256, 256, 0, stream>>>(src, dst, R, K, PK, total);
    };
    cvt(eW1 + (size_t)l * 130 * E1N,  tW1 + (size_t)l * E1NP * EINP, E1N, 130, E1NP, EINP);
    cvt(eW2 + (size_t)l * E1N * HDIM, tW2 + (size_t)l * HDIM * E2KP, HDIM, E1N, HDIM, E2KP);
    cvt(cW1 + (size_t)l * HDIM * CH1, tC1 + (size_t)l * CH1 * HDIM,  CH1, HDIM, CH1, HDIM);
    cvt(nW1 + (size_t)l * 128 * 128,  tN1 + (size_t)l * 128 * NKP,   128, 128, 128, NKP);
    cvt(nW2 + (size_t)l * 128 * HDIM, tN2 + (size_t)l * HDIM * NKP,  HDIM, 128, HDIM, NKP);
  }

  egnn_embed<<<(NN + 255) / 256, 256, 0, stream>>>(x, pos, embedW, embedb, feats, f16, coors);

  for (int l = 0; l < 3; ++l) {
    egnn_zero<<<(NN * HDIM + NN * 3 + 255) / 256, 256, 0, stream>>>(m_acc, c_acc);
    egnn_edge<<<625, WAVES * 32, 0, stream>>>(
        eidx, eattr, coors, f16,
        tW1 + (size_t)l * E1NP * EINP, tW2 + (size_t)l * HDIM * E2KP,
        tC1 + (size_t)l * CH1 * HDIM,
        eb1 + (size_t)l * E1N, eb2 + (size_t)l * HDIM,
        cb1 + (size_t)l * CH1, cW2 + (size_t)l * CH1, cb2 + l,
        m_acc, c_acc);
    egnn_node<<<200, WAVES * 32, 0, stream>>>(
        tN1 + (size_t)l * 128 * NKP, tN2 + (size_t)l * HDIM * NKP,
        nb1 + (size_t)l * 128, nb2 + (size_t)l * HDIM,
        m_acc, c_acc, coors, feats, f16);
  }

  egnn_final<<<(NN + 255) / 256, 256, 0, stream>>>(feats, linW, linb, (float*)d_out);
}
